// ATOCA_58480274703031
// MI455X (gfx1250) — matrix operand order repaired and checked by reading, not compiled, not run on hardware
//
#include <hip/hip_runtime.h>
#include <hip/hip_bf16.h>
#include <math.h>

// ---------------------------------------------------------------------------
// Types for CDNA5 WMMA (wave32): D(16x16 f32) = A(16x32 bf16) x B(32x16 bf16) + C
// ---------------------------------------------------------------------------
typedef __attribute__((ext_vector_type(16))) __bf16 v16bf;
typedef __attribute__((ext_vector_type(8)))  float  v8f;

union FragBF {
    v16bf v;
    unsigned short u16[16];
    unsigned int   u32[8];
    uint4          q[2];
};

__device__ __forceinline__ unsigned short f2bf(float f) {
    unsigned int u = __float_as_uint(f);
    u += 0x7FFFu + ((u >> 16) & 1u);          // round-to-nearest-even
    return (unsigned short)(u >> 16);
}
__device__ __forceinline__ float bf2f(unsigned short h) {
    return __uint_as_float(((unsigned int)h) << 16);
}

// A: row-major bf16 (row = lane&15).  A 16x32 layout (contiguous k, matches B):
//   lanes 0-15: halves 0..15 = K 0..15
//   lanes16-31: halves 0..15 = K 16..31
// B: pre-swizzled fragments of 512 halves: dst[frag*512 + lane*16 + j] = B[kc*32 + (lane>>4)*16 + j][nc*16 + (lane&15)]
__device__ __forceinline__ v8f wmma_loop(const unsigned short* A, int lda,
                                         const unsigned short* Bsw, int NC, int nc,
                                         int kchunks, v8f acc) {
    const int lane = threadIdx.x & 31;
    const int row  = lane & 15;
    const int g    = lane >> 4;
    for (int kc = 0; kc < kchunks; ++kc) {
        FragBF a, b;
        const unsigned short* ap = A + (size_t)row * lda + kc * 32;
        a.q[0] = *(const uint4*)(ap + g * 16);
        a.q[1] = *(const uint4*)(ap + g * 16 + 8);
        const unsigned short* bp = Bsw + (((size_t)(kc * NC + nc)) << 9) + lane * 16;
        b.q[0] = *(const uint4*)(bp);
        b.q[1] = *(const uint4*)(bp + 8);
        acc = __builtin_amdgcn_wmma_f32_16x16x32_bf16(false, a.v, false, b.v,
                                                      (short)0, acc, false, false);
    }
    return acc;
}

// ---------------------------------------------------------------------------
// K0: weight pre-swizzle into B-fragment layout (bf16)
// transposed==0: src[k*N + n]   transposed==1: src[n*K + k]
// ---------------------------------------------------------------------------
__global__ __launch_bounds__(32) void k_swizzleB(const float* __restrict__ src,
                                                 unsigned short* __restrict__ dst,
                                                 int K, int N, int transposed) {
    int frag = blockIdx.x;
    int NC = N >> 4;
    int kc = frag / NC, nc = frag % NC;
    int lane = threadIdx.x;
    int g = lane >> 4;
    int col = nc * 16 + (lane & 15);
    unsigned short* o = dst + ((size_t)frag << 9) + lane * 16;
    for (int j = 0; j < 16; ++j) {
        int k = kc * 32 + g * 16 + j;
        float v = transposed ? src[(size_t)col * K + k] : src[(size_t)k * N + col];
        o[j] = f2bf(v);
    }
}

// ---------------------------------------------------------------------------
// K1/K9: fused per-tile 3x3 conv (im2col in LDS, WMMA 16x192x1728) + GroupNorm
// mode 0 (conv1): gather x via reshape-scramble; epilogue SiLU -> token-major f32
// mode 1 (conv2): gather y token-major;         epilogue GN    -> tile-major  f32
// ---------------------------------------------------------------------------
__global__ __launch_bounds__(128) void k_conv_gn(const float* __restrict__ src,
                                                 const unsigned short* __restrict__ Wsw,
                                                 const float* __restrict__ bias,
                                                 const float* __restrict__ gng,
                                                 const float* __restrict__ gnb,
                                                 float* __restrict__ dst,
                                                 int mode) {
    __shared__ __align__(16) unsigned short sTile[3072];          // [ci*16 + p] bf16
    __shared__ __align__(16) unsigned short sPatch[16 * 1728];    // [pos][ci*9+ky*3+kx]
    __shared__ float sStat[12];
    const int tid  = threadIdx.x;
    const int tile = blockIdx.x;
    const int b = tile / 3136;
    const int t = tile % 3136;

    for (int idx = tid; idx < 3072; idx += 128) {
        int ci = idx >> 4, p = idx & 15;
        float v;
        if (mode == 0) {                       // reshape-scramble of NCHW input
            int s = t * 192 + ci;
            int c = s / 3136, s2 = s % 3136;
            int h = s2 / 14, w = (s2 % 14) * 16 + p;
            v = src[(size_t)(b * 192 + c) * 50176 + h * 224 + w];
        } else {                               // token-major ocab output
            v = src[((size_t)(b * 50176 + t * 16 + p)) * 192 + ci];
        }
        sTile[idx] = f2bf(v);
    }
    __syncthreads();

    for (int idx = tid; idx < 27648; idx += 128) {
        int pos = idx / 1728, k = idx % 1728;
        int ci = k / 9, rr = k - ci * 9;
        int ky = rr / 3, kx = rr - ky * 3;
        int i = (pos >> 2) + ky - 1;
        int j = (pos & 3) + kx - 1;
        unsigned short v = 0;
        if ((unsigned)i < 4u && (unsigned)j < 4u) v = sTile[ci * 16 + i * 4 + j];
        sPatch[pos * 1728 + k] = v;
    }
    __syncthreads();

    const int wave = tid >> 5;
    const int lane = tid & 31;
    const int lg = lane >> 4, ncol = lane & 15;
    v8f acc[3];
    for (int r = 0; r < 3; ++r) {
        int nc = wave + r * 4;
        v8f z = {};
        acc[r] = wmma_loop(sPatch, 1728, Wsw, 12, nc, 54, z);
    }
    __syncthreads();                            // all reads of sPatch done
    float* sOut = (float*)sPatch;               // reuse as 16x192 f32
    for (int r = 0; r < 3; ++r) {
        int nc = wave + r * 4;
        for (int v = 0; v < 8; ++v) {
            int M = v + 8 * lg;
            int col = nc * 16 + ncol;
            sOut[M * 192 + col] = acc[r][v] + bias[col];
        }
    }
    __syncthreads();
    if (tid < 6) {                              // GroupNorm stats: 6 groups x (32ch x 16pos)
        float s = 0.f, ss = 0.f;
        for (int cc = 0; cc < 32; ++cc) {
            int c2 = tid * 32 + cc;
            for (int p = 0; p < 16; ++p) {
                float xx = sOut[p * 192 + c2];
                s += xx; ss += xx * xx;
            }
        }
        float mu = s * (1.f / 512.f);
        float var = ss * (1.f / 512.f) - mu * mu;
        sStat[tid * 2]     = mu;
        sStat[tid * 2 + 1] = rsqrtf(var + 1e-5f);
    }
    __syncthreads();
    for (int idx = tid; idx < 3072; idx += 128) {
        int pos = idx / 192, c2 = idx % 192;
        int g2 = c2 >> 5;
        float xn = (sOut[idx] - sStat[g2 * 2]) * sStat[g2 * 2 + 1] * gng[c2] + gnb[c2];
        if (mode == 0) {
            float si = xn / (1.f + __expf(-xn));                       // SiLU
            dst[((size_t)(b * 50176 + t * 16 + pos)) * 192 + c2] = si; // token-major
        } else {
            dst[(size_t)tile * 3072 + c2 * 16 + pos] = xn;             // tile-major [c'][p]
        }
    }
}

// ---------------------------------------------------------------------------
// LayerNorm: one wave per token (192 ch), bf16 output
// ---------------------------------------------------------------------------
__global__ __launch_bounds__(256) void k_layernorm(const float* __restrict__ in,
                                                   const float* __restrict__ g,
                                                   const float* __restrict__ bta,
                                                   unsigned short* __restrict__ out) {
    int tok = blockIdx.x * 8 + (threadIdx.x >> 5);
    int lane = threadIdx.x & 31;
    const float* row = in + (size_t)tok * 192;
    float v[6];
    float s = 0.f;
    for (int i = 0; i < 6; ++i) { v[i] = row[lane + i * 32]; s += v[i]; }
    for (int o = 16; o; o >>= 1) s += __shfl_xor(s, o, 32);
    float mu = s * (1.f / 192.f);
    float ss = 0.f;
    for (int i = 0; i < 6; ++i) { float d = v[i] - mu; ss += d * d; }
    for (int o = 16; o; o >>= 1) ss += __shfl_xor(ss, o, 32);
    float inv = rsqrtf(ss * (1.f / 192.f) + 1e-5f);
    unsigned short* orow = out + (size_t)tok * 192;
    for (int i = 0; i < 6; ++i) {
        int c = lane + i * 32;
        orow[c] = f2bf((v[i] - mu) * inv * g[c] + bta[c]);
    }
}

// ---------------------------------------------------------------------------
// GEMM stages (one wave = one 16x16 output tile column chunk)
// ---------------------------------------------------------------------------
__global__ __launch_bounds__(128) void k_qkv(const unsigned short* __restrict__ XN,
                                             const unsigned short* __restrict__ Wsw,
                                             const float* __restrict__ bias,
                                             unsigned short* __restrict__ Qg,
                                             unsigned short* __restrict__ Kg,
                                             unsigned short* __restrict__ Vg) {
    int mt = blockIdx.x;
    int nc = blockIdx.y * 4 + (threadIdx.x >> 5);
    int lane = threadIdx.x & 31, lg = lane >> 4, ncol = lane & 15;
    v8f z = {};
    v8f acc = wmma_loop(XN + (size_t)mt * 3072, 192, Wsw, 36, nc, 6, z);
    for (int v = 0; v < 8; ++v) {
        int M = v + 8 * lg;
        int col = nc * 16 + ncol;
        float val = acc[v] + bias[col];
        int tok = mt * 16 + M;
        unsigned short* dst = (col < 192) ? Qg : (col < 384 ? Kg : Vg);
        dst[(size_t)tok * 192 + (col % 192)] = f2bf(val);
    }
}

__global__ __launch_bounds__(128) void k_proj(const unsigned short* __restrict__ AO,
                                              const unsigned short* __restrict__ Wsw,
                                              const float* __restrict__ bias,
                                              const float* __restrict__ X0,
                                              float* __restrict__ OUT) {
    int mt = blockIdx.x;
    int nc = blockIdx.y * 4 + (threadIdx.x >> 5);
    int lane = threadIdx.x & 31, lg = lane >> 4, ncol = lane & 15;
    v8f z = {};
    v8f acc = wmma_loop(AO + (size_t)mt * 3072, 192, Wsw, 12, nc, 6, z);
    for (int v = 0; v < 8; ++v) {
        int M = v + 8 * lg, col = nc * 16 + ncol;
        size_t off = (size_t)(mt * 16 + M) * 192 + col;
        OUT[off] = acc[v] + bias[col] + X0[off];     // + residual shortcut
    }
}

__global__ __launch_bounds__(128) void k_fc1(const unsigned short* __restrict__ XM,
                                             const unsigned short* __restrict__ Wsw,
                                             const float* __restrict__ bias,
                                             unsigned short* __restrict__ Hb) {
    int mt = blockIdx.x;
    int nc = blockIdx.y * 4 + (threadIdx.x >> 5);
    int lane = threadIdx.x & 31, lg = lane >> 4, ncol = lane & 15;
    v8f z = {};
    v8f acc = wmma_loop(XM + (size_t)mt * 3072, 192, Wsw, 24, nc, 6, z);
    for (int v = 0; v < 8; ++v) {
        int M = v + 8 * lg, col = nc * 16 + ncol;
        float val = acc[v] + bias[col];
        float ge = 0.5f * val * (1.f + erff(val * 0.70710678118654752f));  // exact GELU
        Hb[(size_t)(mt * 16 + M) * 384 + col] = f2bf(ge);
    }
}

__global__ __launch_bounds__(128) void k_fc2(const unsigned short* __restrict__ Hb,
                                             const unsigned short* __restrict__ Wsw,
                                             const float* __restrict__ bias,
                                             const float* __restrict__ OUT,
                                             float* __restrict__ Y) {
    int mt = blockIdx.x;
    int nc = blockIdx.y * 4 + (threadIdx.x >> 5);
    int lane = threadIdx.x & 31, lg = lane >> 4, ncol = lane & 15;
    v8f z = {};
    v8f acc = wmma_loop(Hb + (size_t)mt * 16 * 384, 384, Wsw, 12, nc, 12, z);
    for (int v = 0; v < 8; ++v) {
        int M = v + 8 * lg, col = nc * 16 + ncol;
        size_t off = (size_t)(mt * 16 + M) * 192 + col;
        Y[off] = acc[v] + bias[col] + OUT[off];      // + residual
    }
}

__global__ __launch_bounds__(128) void k_conv3(const unsigned short* __restrict__ XPIX,
                                               const unsigned short* __restrict__ Wsw,
                                               const float* __restrict__ bias,
                                               unsigned short* __restrict__ O3) {
    int mt = blockIdx.x;                          // 12544 pixel tiles
    int nc = blockIdx.y * 4 + (threadIdx.x >> 5);
    int lane = threadIdx.x & 31, lg = lane >> 4, ncol = lane & 15;
    v8f z = {};
    v8f acc = wmma_loop(XPIX + (size_t)mt * 3072, 192, Wsw, 12, nc, 6, z);
    for (int v = 0; v < 8; ++v) {
        int M = v + 8 * lg, col = nc * 16 + ncol;
        O3[(size_t)(mt * 16 + M) * 192 + col] = f2bf(acc[v] + bias[col]);
    }
}

// ---------------------------------------------------------------------------
// Attention: one block = one window (16 q x 36 kv), 6 waves = 6 heads
// ---------------------------------------------------------------------------
__global__ __launch_bounds__(192) void k_attn(const unsigned short* __restrict__ Qg,
                                              const unsigned short* __restrict__ Kg,
                                              const unsigned short* __restrict__ Vg,
                                              const float* __restrict__ rpb,
                                              unsigned short* __restrict__ AO) {
    __shared__ __align__(16) unsigned short sQ[16 * 192];
    __shared__ __align__(16) unsigned short sK[36 * 192];
    __shared__ __align__(16) unsigned short sV[36 * 192];
    __shared__ float sS[6 * 16 * 48];
    __shared__ __align__(16) unsigned short sP[6 * 16 * 64];

    const int wI = blockIdx.x;
    const int b  = wI / 3136;
    const int rr = wI % 3136;
    const int nh = rr / 56, nw = rr % 56;
    const int tid = threadIdx.x;

    for (int idx = tid; idx < 3072; idx += 192) {
        int m = idx / 192, c = idx % 192;
        int tok = b * 50176 + (nh * 4 + (m >> 2)) * 224 + nw * 4 + (m & 3);
        sQ[idx] = Qg[(size_t)tok * 192 + c];
    }
    for (int idx = tid; idx < 6912; idx += 192) {
        int e = idx / 192, c = idx % 192;
        int hh = nh * 4 - 1 + e / 6;
        int ww = nw * 4 - 1 + e % 6;
        bool in = ((unsigned)hh < 224u) && ((unsigned)ww < 224u);
        size_t tok = (size_t)b * 50176 + hh * 224 + ww;
        sK[idx] = in ? Kg[tok * 192 + c] : (unsigned short)0;   // zero-padded kv
        sV[idx] = in ? Vg[tok * 192 + c] : (unsigned short)0;
    }
    __syncthreads();

    const int h    = tid >> 5;          // head
    const int lane = tid & 31;
    const int lg   = lane >> 4;
    const int ncol = lane & 15;

    // S = (q * d^-1/2) @ k^T + bias   (3 n-chunks of keys, padded to 48)
    FragBF aq;
    {
        const unsigned short* ap = sQ + ncol * 192 + h * 32;
        aq.q[0] = *(const uint4*)(ap + lg * 16);
        aq.q[1] = *(const uint4*)(ap + lg * 16 + 8);
    }
    for (int nc = 0; nc < 3; ++nc) {
        FragBF bk;
        int kcol = nc * 16 + ncol;
        if (kcol < 36) {
            const unsigned short* kp = sK + kcol * 192 + h * 32 + lg * 16;
            bk.q[0] = *(const uint4*)kp;
            bk.q[1] = *(const uint4*)(kp + 8);
        } else {
            for (int j = 0; j < 8; ++j) bk.u32[j] = 0u;
        }
        v8f acc = {};
        acc = __builtin_amdgcn_wmma_f32_16x16x32_bf16(false, aq.v, false, bk.v,
                                                      (short)0, acc, false, false);
        for (int v = 0; v < 8; ++v) {
            int M = v + 8 * lg;
            int kk = nc * 16 + ncol;
            float sval;
            if (kk < 36) {
                int idx2 = ((kk / 6) - (M >> 2) + 3) * 9 + ((kk % 6) - (M & 3) + 3);
                sval = acc[v] * 0.17677669529663687f + rpb[idx2 * 6 + h];
            } else {
                sval = -1e30f;
            }
            sS[(h * 16 + M) * 48 + kk] = sval;
        }
    }
    __syncthreads();

    if (tid < 96) {                                 // softmax: 1 thread per (head, q)
        float* row = sS + tid * 48;
        float mx = -1e30f;
        for (int k = 0; k < 36; ++k) mx = fmaxf(mx, row[k]);
        float sum = 0.f;
        for (int k = 0; k < 36; ++k) { float e = __expf(row[k] - mx); row[k] = e; sum += e; }
        float inv = 1.f / sum;
        unsigned short* pr = sP + tid * 64;
        for (int k = 0; k < 36; ++k) pr[k] = f2bf(row[k] * inv);
        for (int k = 36; k < 64; ++k) pr[k] = 0;
    }
    __syncthreads();

    // O = P @ V   (K padded to 64, d in 2 chunks of 16)
    for (int nc = 0; nc < 2; ++nc) {
        v8f acc = {};
        for (int kb = 0; kb < 64; kb += 32) {
            FragBF a, bv;
            const unsigned short* ap = sP + (h * 16 + ncol) * 64 + kb;
            a.q[0] = *(const uint4*)(ap + lg * 16);
            a.q[1] = *(const uint4*)(ap + lg * 16 + 8);
            int dcol = nc * 16 + ncol;
            for (int j = 0; j < 16; ++j) {
                int kk = kb + lg * 16 + j;
                bv.u16[j] = (kk < 36) ? sV[kk * 192 + h * 32 + dcol] : (unsigned short)0;
            }
            acc = __builtin_amdgcn_wmma_f32_16x16x32_bf16(false, a.v, false, bv.v,
                                                          (short)0, acc, false, false);
        }
        for (int v = 0; v < 8; ++v) {
            int M = v + 8 * lg;
            int d = nc * 16 + ncol;
            int tok = b * 50176 + (nh * 4 + (M >> 2)) * 224 + nw * 4 + (M & 3);
            AO[(size_t)tok * 192 + h * 32 + d] = f2bf(acc[v]);
        }
    }
}

// ---------------------------------------------------------------------------
// pixel-major bf16 copy of x (for conv3 1x1 GEMM)
// ---------------------------------------------------------------------------
__global__ __launch_bounds__(256) void k_xpix(const float* __restrict__ x,
                                              unsigned short* __restrict__ XPIX) {
    size_t idx = (size_t)blockIdx.x * 256 + threadIdx.x;
    if (idx >= (size_t)200704 * 192) return;
    int p = (int)(idx / 192), c = (int)(idx % 192);
    int b = p / 50176, hw = p % 50176;
    XPIX[idx] = f2bf(x[((size_t)(b * 192 + c)) * 50176 + hw]);
}

// ---------------------------------------------------------------------------
// final: out[b,co,h,w] = relu(gn2(reinterpreted) + conv3)
// ---------------------------------------------------------------------------
__global__ __launch_bounds__(256) void k_final(const float* __restrict__ GN2,
                                               const unsigned short* __restrict__ O3,
                                               float* __restrict__ out) {
    size_t idx = (size_t)blockIdx.x * 256 + threadIdx.x;
    if (idx >= (size_t)2 * 192 * 50176) return;
    int b = (int)(idx / 9633792);
    int f = (int)(idx % 9633792);
    int co = f / 50176;
    int rest = f % 50176;
    int h = rest / 224, w = rest % 224;
    int sp = h * 14 + (w >> 4);
    int p  = w & 15;
    int s  = co * 3136 + sp;
    int t  = s / 192, c2 = s % 192;
    float g2 = GN2[((size_t)(b * 3136 + t)) * 3072 + c2 * 16 + p];
    float o3 = bf2f(O3[((size_t)(b * 50176 + h * 224 + w)) * 192 + co]);
    float r = g2 + o3;
    out[idx] = r > 0.f ? r : 0.f;
}

// ---------------------------------------------------------------------------
// Workspace layout (regions recycled by lifetime)
// ---------------------------------------------------------------------------
#define OFF_W 0                                   // weights bf16 (~2MB)
#define OFF_A 2097152                             // 77MB: X0 -> Y -> XPIX
#define OFF_B (OFF_A + 77070336)                  // 38.5MB: XN -> AO -> XM
#define OFF_C (OFF_B + 38535168)                  // 115.6MB: Q,K,V -> Hb -> O3
#define OFF_D (OFF_C + 115605504)                 // 77MB: OUT -> GN2

extern "C" void kernel_launch(void* const* d_in, const int* in_sizes, int n_in,
                              void* d_out, int out_size, void* d_ws, size_t ws_size,
                              hipStream_t stream) {
    const float* x       = (const float*)d_in[0];
    const float* conv1_w = (const float*)d_in[1];
    const float* conv1_b = (const float*)d_in[2];
    const float* gn1_g   = (const float*)d_in[3];
    const float* gn1_b   = (const float*)d_in[4];
    const float* conv2_w = (const float*)d_in[5];
    const float* conv2_b = (const float*)d_in[6];
    const float* gn2_g   = (const float*)d_in[7];
    const float* gn2_b   = (const float*)d_in[8];
    const float* conv3_w = (const float*)d_in[9];
    const float* conv3_b = (const float*)d_in[10];
    const float* ln1_g   = (const float*)d_in[11];
    const float* ln1_b   = (const float*)d_in[12];
    const float* qkv_w   = (const float*)d_in[13];
    const float* qkv_b   = (const float*)d_in[14];
    const float* rpb     = (const float*)d_in[15];
    const float* proj_w  = (const float*)d_in[16];
    const float* proj_b  = (const float*)d_in[17];
    const float* ln2_g   = (const float*)d_in[18];
    const float* ln2_b   = (const float*)d_in[19];
    const float* fc1_w   = (const float*)d_in[20];
    const float* fc1_b   = (const float*)d_in[21];
    const float* fc2_w   = (const float*)d_in[22];
    const float* fc2_b   = (const float*)d_in[23];

    char* ws = (char*)d_ws;
    unsigned short* wb   = (unsigned short*)(ws + OFF_W);
    unsigned short* Wc1  = wb + 0;        // 54*12*512
    unsigned short* Wqkv = wb + 331776;   // 6*36*512
    unsigned short* Wprj = wb + 442368;   // 6*12*512
    unsigned short* Wf1  = wb + 479232;   // 6*24*512
    unsigned short* Wf2  = wb + 552960;   // 12*12*512
    unsigned short* Wc2  = wb + 626688;   // 54*12*512
    unsigned short* Ww3  = wb + 958464;   // 6*12*512

    float*          X0   = (float*)(ws + OFF_A);
    unsigned short* XN   = (unsigned short*)(ws + OFF_B);
    unsigned short* Qg   = (unsigned short*)(ws + OFF_C);
    unsigned short* Kg   = Qg + 19267584;
    unsigned short* Vg   = Qg + 38535168;
    unsigned short* AO   = (unsigned short*)(ws + OFF_B);
    float*          OUT  = (float*)(ws + OFF_D);
    unsigned short* XM   = (unsigned short*)(ws + OFF_B);
    unsigned short* Hb   = (unsigned short*)(ws + OFF_C);
    float*          Y    = (float*)(ws + OFF_A);
    float*          GN2  = (float*)(ws + OFF_D);
    unsigned short* XPIX = (unsigned short*)(ws + OFF_A);
    unsigned short* O3   = (unsigned short*)(ws + OFF_C);

    // weight pre-swizzle (bf16 B-fragments)
    k_swizzleB<<<54 * 12, 32, 0, stream>>>(conv1_w, Wc1, 1728, 192, 1);
    k_swizzleB<<<6 * 36, 32, 0, stream>>>(qkv_w, Wqkv, 192, 576, 0);
    k_swizzleB<<<6 * 12, 32, 0, stream>>>(proj_w, Wprj, 192, 192, 0);
    k_swizzleB<<<6 * 24, 32, 0, stream>>>(fc1_w, Wf1, 192, 384, 0);
    k_swizzleB<<<12 * 12, 32, 0, stream>>>(fc2_w, Wf2, 384, 192, 0);
    k_swizzleB<<<54 * 12, 32, 0, stream>>>(conv2_w, Wc2, 1728, 192, 1);
    k_swizzleB<<<6 * 12, 32, 0, stream>>>(conv3_w, Ww3, 192, 192, 1);

    k_conv_gn<<<6272, 128, 0, stream>>>(x, Wc1, conv1_b, gn1_g, gn1_b, X0, 0);
    k_layernorm<<<12544, 256, 0, stream>>>(X0, ln1_g, ln1_b, XN);
    k_qkv<<<dim3(6272, 9), 128, 0, stream>>>(XN, Wqkv, qkv_b, Qg, Kg, Vg);
    k_attn<<<6272, 192, 0, stream>>>(Qg, Kg, Vg, rpb, AO);
    k_proj<<<dim3(6272, 3), 128, 0, stream>>>(AO, Wprj, proj_b, X0, OUT);
    k_layernorm<<<12544, 256, 0, stream>>>(OUT, ln2_g, ln2_b, XM);
    k_fc1<<<dim3(6272, 6), 128, 0, stream>>>(XM, Wf1, fc1_b, Hb);
    k_fc2<<<dim3(6272, 3), 128, 0, stream>>>(Hb, Wf2, fc2_b, OUT, Y);
    k_conv_gn<<<6272, 128, 0, stream>>>(Y, Wc2, conv2_b, gn2_g, gn2_b, GN2, 1);
    k_xpix<<<150528, 256, 0, stream>>>(x, XPIX);
    k_conv3<<<dim3(12544, 3), 128, 0, stream>>>(XPIX, Ww3, conv3_b, O3);
    k_final<<<75264, 256, 0, stream>>>(GN2, O3, (float*)d_out);
}
